// MoEMambaTransformerBlock_6803228197408
// MI455X (gfx1250) — matrix operand order repaired and checked by reading, not compiled, not run on hardware
//
#include <hip/hip_runtime.h>
#include <math.h>

#define D_MODEL 512
#define NHEAD 8
#define HD 64
#define D_INNER 1024
#define HM 16
#define PP 64
#define NN 64
#define D_CONV 4
#define CONV_DIM (D_INNER + 2*NN)          // 1152
#define D_IN_PROJ (2*D_INNER + 2*NN + HM)  // 2192
#define D_FF (4*D_MODEL)                   // 2048
#define BB 4
#define TT 1024
#define SS 1024
#define EPSF 1e-5f

typedef __attribute__((ext_vector_type(16))) __bf16 v16bf;
typedef __attribute__((ext_vector_type(8)))  float  v8f;

union BF16x16 { uint4 q[2]; v16bf v; unsigned short u[16]; };
union F16x4   { float4 f4[4]; float f[16]; };

__device__ __forceinline__ unsigned short f2bf(float f) {
  union { float f; unsigned u; } x; x.f = f;
  unsigned r = x.u + 0x7FFFu + ((x.u >> 16) & 1u);  // round-to-nearest-even
  return (unsigned short)(r >> 16);
}
__device__ __forceinline__ unsigned pack2bf(float a, float b) {
  return (unsigned)f2bf(a) | ((unsigned)f2bf(b) << 16);
}
__device__ __forceinline__ float siluf(float v) { return v / (1.f + expf(-v)); }
__device__ __forceinline__ float geluf(float v) {
  const float c = 0.7978845608028654f;
  return 0.5f * v * (1.f + tanhf(c * (v + 0.044715f * v * v * v)));
}

// ---------------------------------------------------------------- LayerNorm
__global__ void layernorm_k(const float* __restrict__ X, const float* __restrict__ w,
                            const float* __restrict__ b, float* __restrict__ Y, int D) {
  __shared__ float red[256];
  int t = blockIdx.x, tid = threadIdx.x;
  const float* xr = X + (size_t)t * D;
  float s = 0.f;
  for (int i = tid; i < D; i += 256) s += xr[i];
  red[tid] = s; __syncthreads();
  for (int o = 128; o > 0; o >>= 1) { if (tid < o) red[tid] += red[tid + o]; __syncthreads(); }
  float mean = red[0] / D;
  __syncthreads();
  float v = 0.f;
  for (int i = tid; i < D; i += 256) { float d = xr[i] - mean; v += d * d; }
  red[tid] = v; __syncthreads();
  for (int o = 128; o > 0; o >>= 1) { if (tid < o) red[tid] += red[tid + o]; __syncthreads(); }
  float rstd = rsqrtf(red[0] / D + EPSF);
  for (int i = tid; i < D; i += 256)
    Y[(size_t)t * D + i] = (xr[i] - mean) * rstd * w[i] + b[i];
}

// ---------------------------------------------------------------- WMMA GEMM
// C[M,N] = act(A[M,K] @ B[K,N] + bias) + resid.  Requirements honored by all call
// sites: K multiple of 32.  M-edge / N-edge handled branch-free with clamped
// addresses + multiplicative masks precomputed once per thread; all staging
// swizzle math is hoisted out of the K-loop (only pointers advance).
#define BM 32
#define BN 128
#define BK 32
__global__ void gemm_wmma(const float* __restrict__ A, const float* __restrict__ Bw,
                          const float* __restrict__ bias, const float* __restrict__ resid,
                          float* __restrict__ C, int M, int N, int K, int act) {
  __shared__ __align__(16) unsigned int AfragU[(BM * BK) / 2];   // 512 u32 = 2KB
  __shared__ __align__(16) unsigned int BfragU[(BK * BN) / 2];   // 2048 u32 = 8KB
  int tid = threadIdx.x;
  int lane = tid & 31;
  int wave = tid >> 5;              // 0..7
  int wr = wave >> 2, wc = wave & 3;
  int bm = blockIdx.y * BM, bn = blockIdx.x * BN;

  // ---- precompute per-thread staging slots (loop-invariant)
  unsigned* aDst[2]; const float* aSrc[2]; float aMask[2];
#pragma unroll
  for (int j = 0; j < 2; ++j) {
    int i = tid + j * 256;
    int r = i >> 4, c = (i & 15) * 2;
    int gr = bm + r;
    aMask[j] = (gr < M) ? 1.f : 0.f;
    int grc = (gr < M) ? gr : (M - 1);
    int wrr = r >> 4, m = r & 15;
    int half = (c >> 3) & 1;
    int e = (c & 7) + ((c >= 16) ? 8 : 0);       // even
    aDst[j] = &AfragU[(wrr * 32 + half * 16 + m) * 8 + (e >> 1)];
    aSrc[j] = A + (size_t)grc * K + c;
  }
  unsigned* bDst[8]; const float* bSrc[8]; float bMask[8];
#pragma unroll
  for (int j = 0; j < 8; ++j) {
    int i = tid + j * 256;
    int kr = (i >> 7) * 2, ncol = i & 127;
    int gc = bn + ncol;
    bMask[j] = (gc < N) ? 1.f : 0.f;
    int gcc = (gc < N) ? gc : (N - 1);
    int sub = ncol >> 4, n = ncol & 15;
    int half = (kr >> 3) & 1, e = (kr & 7) + ((kr >= 16) ? 8 : 0);   // even
    bDst[j] = &BfragU[(sub * 32 + half * 16 + n) * 8 + (e >> 1)];
    bSrc[j] = Bw + (size_t)kr * N + gcc;
  }

  v8f acc0 = {}, acc1 = {};
  int nk = K / BK;
  for (int kb = 0; kb < nk; ++kb) {
    // speculative prefetch of next tiles (CDNA5 global_prefetch_b8)
    __builtin_prefetch(aSrc[0] + BK, 0, 0);
    __builtin_prefetch(bSrc[0] + (size_t)BK * N, 0, 0);
#pragma unroll
    for (int j = 0; j < 2; ++j) {
      float2 t = *(const float2*)aSrc[j];
      *aDst[j] = pack2bf(t.x * aMask[j], t.y * aMask[j]);
      aSrc[j] += BK;
    }
#pragma unroll
    for (int j = 0; j < 8; ++j) {
      float v0 = bSrc[j][0] * bMask[j];
      float v1 = bSrc[j][N] * bMask[j];
      *bDst[j] = pack2bf(v0, v1);
      bSrc[j] += (size_t)BK * N;
    }
    __syncthreads();
    BF16x16 af, b0, b1;
    const uint4* ap  = (const uint4*)&AfragU[(wr * 32 + lane) * 8];
    af.q[0] = ap[0]; af.q[1] = ap[1];
    const uint4* bp0 = (const uint4*)&BfragU[((wc * 2 + 0) * 32 + lane) * 8];
    b0.q[0] = bp0[0]; b0.q[1] = bp0[1];
    const uint4* bp1 = (const uint4*)&BfragU[((wc * 2 + 1) * 32 + lane) * 8];
    b1.q[0] = bp1[0]; b1.q[1] = bp1[1];
    acc0 = __builtin_amdgcn_wmma_f32_16x16x32_bf16(false, af.v, false, b0.v,
                                                   (short)0, acc0, false, false);
    acc1 = __builtin_amdgcn_wmma_f32_16x16x32_bf16(false, af.v, false, b1.v,
                                                   (short)0, acc1, false, false);
    __syncthreads();
  }
  int rowbase = bm + wr * 16 + ((lane < 16) ? 0 : 8);
#pragma unroll
  for (int j = 0; j < 2; ++j) {
    int col = bn + wc * 32 + j * 16 + (lane & 15);
    v8f acc = j ? acc1 : acc0;
#pragma unroll
    for (int v = 0; v < 8; ++v) {
      int row = rowbase + v;
      if (row < M && col < N) {
        float val = acc[v];
        if (bias) val += bias[col];
        if (act == 1) val = geluf(val);
        if (resid) val += resid[(size_t)row * N + col];
        C[(size_t)row * N + col] = val;
      }
    }
  }
}

// ------------------------------------------------------- Flash attention (WMMA)
// One wave per 16 query rows.  Assumes Tq, Tkv multiples of 32 (true here:
// 1024), so no key-range guards needed; causal masking only on the diagonal.
__global__ void flash_attn(const float* __restrict__ Q, const float* __restrict__ K,
                           const float* __restrict__ V, float* __restrict__ Out,
                           int sq, int skv, int Tq, int Tkv, int causal,
                           float scale, int nhead) {
  __shared__ __align__(16) unsigned short pLds[16 * 32];        // 1KB, A-frag order
  __shared__ __align__(16) unsigned int   vLds[4 * 32 * 8];     // 4KB, B-frag order
  int lane = threadIdx.x & 31;
  int qt = blockIdx.x;
  int b = blockIdx.y / nhead, h = blockIdx.y % nhead;
  const float* Qb = Q + (size_t)b * Tq * sq + h * HD;
  const float* Kb = K + (size_t)b * Tkv * skv + h * HD;
  const float* Vb = V + (size_t)b * Tkv * skv + h * HD;

  int m = lane & 15;
  int kbase = (lane < 16) ? 0 : 8;     // A-layout K base
  int nlane = lane & 15;
  int kcbase = (lane < 16) ? 0 : 8;    // B-layout K base
  int rowoff = (lane < 16) ? 0 : 8;    // C-layout row offset

  // Q tile in A-layout: two K=32 chunks covering d 0..63; float4 loads
  BF16x16 q0, q1;
  {
    int qrowA = qt * 16 + m;
    const float* qr = Qb + (size_t)qrowA * sq;
    F16x4 t0, t1;
    t0.f4[0] = *(const float4*)(qr + kbase);
    t0.f4[1] = *(const float4*)(qr + kbase + 4);
    t0.f4[2] = *(const float4*)(qr + kbase + 16);
    t0.f4[3] = *(const float4*)(qr + kbase + 20);
    t1.f4[0] = *(const float4*)(qr + 32 + kbase);
    t1.f4[1] = *(const float4*)(qr + 32 + kbase + 4);
    t1.f4[2] = *(const float4*)(qr + 32 + kbase + 16);
    t1.f4[3] = *(const float4*)(qr + 32 + kbase + 20);
#pragma unroll
    for (int e = 0; e < 16; ++e) { q0.u[e] = f2bf(t0.f[e]); q1.u[e] = f2bf(t1.f[e]); }
  }

  v8f o[4] = {};
  float rmax[8], rsum[8];
#pragma unroll
  for (int v = 0; v < 8; ++v) { rmax[v] = -1e30f; rsum[v] = 0.f; }

  int qmaxrow = qt * 16 + 15;
  int kend = causal ? ((qmaxrow + 1 < Tkv) ? qmaxrow + 1 : Tkv) : Tkv;
  int nblk = (kend + 31) / 32;

  for (int jb = 0; jb < nblk; ++jb) {
    int kb0 = jb * 32;
    // ---- S = Q @ K^T for 32 keys (two 16-key subtiles); keys always in range
    v8f s[2];
#pragma unroll
    for (int st = 0; st < 2; ++st) {
      int keyA = kb0 + st * 16 + nlane;
      const float* kr = Kb + (size_t)keyA * skv;
      BF16x16 bk0, bk1;
      F16x4 t0, t1;
#pragma unroll
      for (int qq = 0; qq < 4; ++qq) {
        t0.f4[qq] = *(const float4*)(kr + kcbase + qq * 4 + (qq >> 1) * 8);
        t1.f4[qq] = *(const float4*)(kr + 32 + kcbase + qq * 4 + (qq >> 1) * 8);
      }
#pragma unroll
      for (int e = 0; e < 16; ++e) { bk0.u[e] = f2bf(t0.f[e]); bk1.u[e] = f2bf(t1.f[e]); }
      v8f sa = {};
      sa = __builtin_amdgcn_wmma_f32_16x16x32_bf16(false, q0.v, false, bk0.v,
                                                   (short)0, sa, false, false);
      sa = __builtin_amdgcn_wmma_f32_16x16x32_bf16(false, q1.v, false, bk1.v,
                                                   (short)0, sa, false, false);
      s[st] = sa;
    }
    // ---- scale + causal mask + row max
    float tmax[8];
#pragma unroll
    for (int v = 0; v < 8; ++v) tmax[v] = -1e30f;
#pragma unroll
    for (int st = 0; st < 2; ++st) {
      int key = kb0 + st * 16 + nlane;
#pragma unroll
      for (int v = 0; v < 8; ++v) {
        float sv = s[st][v] * scale;
        int qrow = qt * 16 + v + rowoff;
        if (causal && key > qrow) sv = -1e30f;
        s[st][v] = sv;
        tmax[v] = fmaxf(tmax[v], sv);
      }
    }
#pragma unroll
    for (int v = 0; v < 8; ++v) {
      float t = tmax[v];
      t = fmaxf(t, __shfl_xor(t, 1, 32));
      t = fmaxf(t, __shfl_xor(t, 2, 32));
      t = fmaxf(t, __shfl_xor(t, 4, 32));
      t = fmaxf(t, __shfl_xor(t, 8, 32));
      tmax[v] = t;
    }
    // ---- online softmax
    float corr[8], psum[8];
#pragma unroll
    for (int v = 0; v < 8; ++v) {
      float mnew = fmaxf(rmax[v], tmax[v]);
      corr[v] = expf(rmax[v] - mnew);
      rmax[v] = mnew;
      psum[v] = 0.f;
    }
#pragma unroll
    for (int st = 0; st < 2; ++st)
#pragma unroll
      for (int v = 0; v < 8; ++v) {
        float p = expf(s[st][v] - rmax[v]);
        s[st][v] = p;
        psum[v] += p;
      }
#pragma unroll
    for (int v = 0; v < 8; ++v) {
      float t = psum[v];
      t += __shfl_xor(t, 1, 32);
      t += __shfl_xor(t, 2, 32);
      t += __shfl_xor(t, 4, 32);
      t += __shfl_xor(t, 8, 32);
      rsum[v] = rsum[v] * corr[v] + t;
    }
#pragma unroll
    for (int dt = 0; dt < 4; ++dt)
#pragma unroll
      for (int v = 0; v < 8; ++v) o[dt][v] *= corr[v];

    __syncthreads();
    // ---- P (C-layout f32) -> LDS in A-fragment order (bf16)
#pragma unroll
    for (int st = 0; st < 2; ++st)
#pragma unroll
      for (int v = 0; v < 8; ++v) {
        int row = v + rowoff;
        int k = st * 16 + nlane;
        int half = (k >> 3) & 1;
        int e = (k & 7) + ((k >= 16) ? 8 : 0);
        pLds[(half * 16 + row) * 16 + e] = f2bf(s[st][v]);
      }
    // ---- V block (32 keys x 64 d) -> LDS in B-fragment order, paired keys
    {
      int g = lane & 15, dhalf = lane >> 4;
      int key0 = kb0 + 2 * g;
      int half = (g >> 2) & 1;         // = ((2g)>>3)&1
      int e = ((2 * g) & 7) + ((g >= 8) ? 8 : 0);   // even
      const float* v0p = Vb + (size_t)key0 * skv;
      const float* v1p = v0p + skv;
#pragma unroll
      for (int dq = 0; dq < 8; ++dq) {
        int d0 = dhalf * 32 + dq * 4;
        float4 a = *(const float4*)(v0p + d0);
        float4 c = *(const float4*)(v1p + d0);
        float av[4] = {a.x, a.y, a.z, a.w};
        float cv[4] = {c.x, c.y, c.z, c.w};
#pragma unroll
        for (int cc = 0; cc < 4; ++cc) {
          int d = d0 + cc;
          int sub = d >> 4, n = d & 15;
          vLds[(sub * 32 + half * 16 + n) * 8 + (e >> 1)] = pack2bf(av[cc], cv[cc]);
        }
      }
    }
    __syncthreads();
    // ---- O += P @ V  (all operands via ds_load_b128)
    BF16x16 pa;
    const uint4* pp = (const uint4*)&pLds[lane * 16];
    pa.q[0] = pp[0]; pa.q[1] = pp[1];
#pragma unroll
    for (int dt = 0; dt < 4; ++dt) {
      BF16x16 bv;
      const uint4* vp = (const uint4*)&vLds[(dt * 32 + lane) * 8];
      bv.q[0] = vp[0]; bv.q[1] = vp[1];
      o[dt] = __builtin_amdgcn_wmma_f32_16x16x32_bf16(false, pa.v, false, bv.v,
                                                      (short)0, o[dt], false, false);
    }
  }
  float* Ob = Out + (size_t)b * Tq * (nhead * HD) + h * HD;
#pragma unroll
  for (int v = 0; v < 8; ++v) {
    int qrow = qt * 16 + v + rowoff;
    float inv = 1.f / rsum[v];
    float* orow = Ob + (size_t)qrow * (nhead * HD);
    orow[ 0 + nlane] = o[0][v] * inv;
    orow[16 + nlane] = o[1][v] * inv;
    orow[32 + nlane] = o[2][v] * inv;
    orow[48 + nlane] = o[3][v] * inv;
  }
}

// ----------------------------------------------------- Mamba: conv1d + SiLU
__global__ void conv_silu_k(const float* __restrict__ zx, const float* __restrict__ cw,
                            const float* __restrict__ cb, float* __restrict__ out) {
  int token = blockIdx.x;
  int b = token / TT, t = token % TT;
  for (int c = threadIdx.x; c < CONV_DIM; c += 256) {
    float acc = cb[c];
#pragma unroll
    for (int k = 0; k < D_CONV; ++k) {
      int tt = t - (D_CONV - 1) + k;
      if (tt >= 0)
        acc += zx[(size_t)(b * TT + tt) * D_IN_PROJ + D_INNER + c] * cw[c * D_CONV + k];
    }
    out[(size_t)token * CONV_DIM + c] = siluf(acc);
  }
}

// ----------------------------------------------------- Mamba: dt / dA precompute
__global__ void dt_k(const float* __restrict__ zx, const float* __restrict__ dt_bias,
                     const float* __restrict__ A_log, float* __restrict__ dtv,
                     float* __restrict__ dAv, int n) {
  int i = blockIdx.x * 256 + threadIdx.x;
  if (i >= n) return;
  int h = i % HM;
  int token = i / HM;
  float raw = zx[(size_t)token * D_IN_PROJ + D_INNER + CONV_DIM + h] + dt_bias[h];
  float sp = (raw > 20.f) ? raw : log1pf(expf(raw));
  dtv[i] = sp;
  dAv[i] = expf(-expf(A_log[h]) * sp);
}

// ----------------------------------------------------- Mamba: selective scan
__global__ void scan_k(const float* __restrict__ xbc, const float* __restrict__ dtv,
                       const float* __restrict__ dAv, const float* __restrict__ Dskip,
                       float* __restrict__ ym) {
  __shared__ float sx[PP], sB[NN], sC[NN];
  int bh = blockIdx.x;
  int b = bh / HM, hh = bh % HM;
  int tid = threadIdx.x;
  int p = tid >> 2, nq = tid & 3;
  float hst[16];
#pragma unroll
  for (int i = 0; i < 16; ++i) hst[i] = 0.f;
  float dsk = Dskip[hh];
  for (int t = 0; t < TT; ++t) {
    int token = b * TT + t;
    __syncthreads();
    if (tid < 64)        sx[tid]       = xbc[(size_t)token * CONV_DIM + hh * PP + tid];
    else if (tid < 128)  sB[tid - 64]  = xbc[(size_t)token * CONV_DIM + D_INNER + (tid - 64)];
    else if (tid < 192)  sC[tid - 128] = xbc[(size_t)token * CONV_DIM + D_INNER + NN + (tid - 128)];
    __syncthreads();
    float da = dAv[(size_t)token * HM + hh];
    float dt = dtv[(size_t)token * HM + hh];
    float xp = sx[p];
    float coef = dt * xp;
    float acc = 0.f;
#pragma unroll
    for (int i = 0; i < 16; ++i) {
      int n = nq * 16 + i;
      hst[i] = hst[i] * da + coef * sB[n];
      acc += hst[i] * sC[n];
    }
    acc += __shfl_xor(acc, 1, 32);
    acc += __shfl_xor(acc, 2, 32);
    if (nq == 0)
      ym[(size_t)token * D_INNER + hh * PP + p] = acc + dsk * xp;
  }
}

// ----------------------------------------------------- Mamba: gate + RMSNorm
__global__ void gate_rms_k(const float* __restrict__ ym, const float* __restrict__ zx,
                           const float* __restrict__ nw, float* __restrict__ yn) {
  __shared__ float red[256];
  int token = blockIdx.x, tid = threadIdx.x;
  float vals[4];
  float ss = 0.f;
#pragma unroll
  for (int j = 0; j < 4; ++j) {
    int i = tid + j * 256;
    float z = zx[(size_t)token * D_IN_PROJ + i];
    float y = ym[(size_t)token * D_INNER + i] * siluf(z);
    vals[j] = y;
    ss += y * y;
  }
  red[tid] = ss; __syncthreads();
  for (int o = 128; o > 0; o >>= 1) { if (tid < o) red[tid] += red[tid + o]; __syncthreads(); }
  float scale = rsqrtf(red[0] / D_INNER + EPSF);
#pragma unroll
  for (int j = 0; j < 4; ++j) {
    int i = tid + j * 256;
    yn[(size_t)token * D_INNER + i] = vals[j] * scale * nw[i];
  }
}

// ----------------------------------------------------- Router softmax + combine
__global__ void router_k(const float* __restrict__ x, const float* __restrict__ nx,
                         const float* __restrict__ rw, const float* __restrict__ rb,
                         const float* __restrict__ om, const float* __restrict__ oa,
                         float* __restrict__ x1) {
  __shared__ float r0s[256], r1s[256];
  int token = blockIdx.x, tid = threadIdx.x;
  float d0 = 0.f, d1 = 0.f;
  for (int i = tid; i < D_MODEL; i += 256) {
    float v = nx[(size_t)token * D_MODEL + i];
    d0 += v * rw[i * 2 + 0];
    d1 += v * rw[i * 2 + 1];
  }
  r0s[tid] = d0; r1s[tid] = d1; __syncthreads();
  for (int o = 128; o > 0; o >>= 1) {
    if (tid < o) { r0s[tid] += r0s[tid + o]; r1s[tid] += r1s[tid + o]; }
    __syncthreads();
  }
  float l0 = r0s[0] + rb[0], l1 = r1s[0] + rb[1];
  float mx = fmaxf(l0, l1);
  float e0 = expf(l0 - mx), e1 = expf(l1 - mx);
  float inv = 1.f / (e0 + e1);
  float w0 = e0 * inv, w1 = e1 * inv;
  for (int i = tid; i < D_MODEL; i += 256) {
    size_t idx = (size_t)token * D_MODEL + i;
    x1[idx] = x[idx] + w0 * om[idx] + w1 * oa[idx];
  }
}

// ==================================================================== host
extern "C" void kernel_launch(void* const* d_in, const int* in_sizes, int n_in,
                              void* d_out, int out_size, void* d_ws, size_t ws_size,
                              hipStream_t stream) {
  (void)in_sizes; (void)n_in; (void)out_size; (void)ws_size;
  const float* x          = (const float*)d_in[0];
  const float* memory     = (const float*)d_in[1];
  const float* in_proj_w  = (const float*)d_in[2];
  const float* conv_w     = (const float*)d_in[3];
  const float* conv_b     = (const float*)d_in[4];
  const float* dt_bias    = (const float*)d_in[5];
  const float* A_log      = (const float*)d_in[6];
  const float* D_skip     = (const float*)d_in[7];
  const float* mamba_norm_w = (const float*)d_in[8];
  const float* mamba_out_w  = (const float*)d_in[9];
  const float* attn_in_w  = (const float*)d_in[10];
  const float* attn_in_b  = (const float*)d_in[11];
  const float* attn_out_w = (const float*)d_in[12];
  const float* attn_out_b = (const float*)d_in[13];
  const float* router_w   = (const float*)d_in[14];
  const float* router_b   = (const float*)d_in[15];
  const float* q_w        = (const float*)d_in[16];
  const float* kv_w       = (const float*)d_in[17];
  const float* co_w       = (const float*)d_in[18];
  const float* co_b       = (const float*)d_in[19];
  const float* ln1_w      = (const float*)d_in[20];
  const float* ln1_b      = (const float*)d_in[21];
  const float* ln2_w      = (const float*)d_in[22];
  const float* ln2_b      = (const float*)d_in[23];
  const float* ln3_w      = (const float*)d_in[24];
  const float* ln3_b      = (const float*)d_in[25];
  const float* ffn_w1     = (const float*)d_in[26];
  const float* ffn_b1     = (const float*)d_in[27];
  const float* ffn_w2     = (const float*)d_in[28];
  const float* ffn_b2     = (const float*)d_in[29];

  const int M = BB * TT;  // 4096 tokens
  char* wsb = (char*)d_ws;
  size_t off = 0;
  auto alloc = [&](size_t elems) -> float* {
    float* p = (float*)(wsb + off);
    off += ((elems * sizeof(float) + 255) / 256) * 256;
    return p;
  };
  float* nx  = alloc((size_t)M * D_MODEL);
  float* zx  = alloc((size_t)M * D_IN_PROJ);
  float* xbc = alloc((size_t)M * CONV_DIM);
  float* dtv = alloc((size_t)M * HM);
  float* dAv = alloc((size_t)M * HM);
  float* ym  = alloc((size_t)M * D_INNER);
  float* yn  = alloc((size_t)M * D_INNER);
  float* om  = alloc((size_t)M * D_MODEL);
  float* qkv = alloc((size_t)M * 3 * D_MODEL);
  float* am  = alloc((size_t)M * D_MODEL);
  float* oa  = alloc((size_t)M * D_MODEL);
  float* x1  = alloc((size_t)M * D_MODEL);
  float* n2  = alloc((size_t)M * D_MODEL);
  float* qc  = alloc((size_t)M * D_MODEL);
  float* kv  = alloc((size_t)BB * SS * 2 * D_MODEL);
  float* ca  = alloc((size_t)M * D_MODEL);
  float* x2  = alloc((size_t)M * D_MODEL);
  float* n3  = alloc((size_t)M * D_MODEL);
  float* ff  = alloc((size_t)M * D_FF);

  dim3 blk256(256);
  auto gemmL = [&](const float* A, const float* Bw, const float* bias, const float* resid,
                   float* C, int Mm, int Nn, int Kk, int act) {
    dim3 g((Nn + BN - 1) / BN, (Mm + BM - 1) / BM);
    gemm_wmma<<<g, blk256, 0, stream>>>(A, Bw, bias, resid, C, Mm, Nn, Kk, act);
  };

  // ---- pre-norm
  layernorm_k<<<M, blk256, 0, stream>>>(x, ln1_w, ln1_b, nx, D_MODEL);

  // ---- Mamba branch
  gemmL(nx, in_proj_w, nullptr, nullptr, zx, M, D_IN_PROJ, D_MODEL, 0);
  conv_silu_k<<<M, blk256, 0, stream>>>(zx, conv_w, conv_b, xbc);
  int ndt = M * HM;
  dt_k<<<(ndt + 255) / 256, blk256, 0, stream>>>(zx, dt_bias, A_log, dtv, dAv, ndt);
  scan_k<<<BB * HM, blk256, 0, stream>>>(xbc, dtv, dAv, D_skip, ym);
  gate_rms_k<<<M, blk256, 0, stream>>>(ym, zx, mamba_norm_w, yn);
  gemmL(yn, mamba_out_w, nullptr, nullptr, om, M, D_MODEL, D_INNER, 0);

  // ---- causal self-attention branch
  gemmL(nx, attn_in_w, attn_in_b, nullptr, qkv, M, 3 * D_MODEL, D_MODEL, 0);
  {
    dim3 g(TT / 16, BB * NHEAD); dim3 bb(32);
    flash_attn<<<g, bb, 0, stream>>>(qkv, qkv + D_MODEL, qkv + 2 * D_MODEL, am,
                                     3 * D_MODEL, 3 * D_MODEL, TT, TT, 1, 0.125f, NHEAD);
  }
  gemmL(am, attn_out_w, attn_out_b, nullptr, oa, M, D_MODEL, D_MODEL, 0);

  // ---- router mix + residual
  router_k<<<M, blk256, 0, stream>>>(x, nx, router_w, router_b, om, oa, x1);

  // ---- cross-attention
  layernorm_k<<<M, blk256, 0, stream>>>(x1, ln2_w, ln2_b, n2, D_MODEL);
  gemmL(n2, q_w, nullptr, nullptr, qc, M, D_MODEL, D_MODEL, 0);
  gemmL(memory, kv_w, nullptr, nullptr, kv, BB * SS, 2 * D_MODEL, D_MODEL, 0);
  {
    dim3 g(TT / 16, BB * NHEAD); dim3 bb(32);
    flash_attn<<<g, bb, 0, stream>>>(qc, kv, kv + D_MODEL, ca,
                                     D_MODEL, 2 * D_MODEL, TT, SS, 0, 0.125f, NHEAD);
  }
  gemmL(ca, co_w, co_b, x1, x2, M, D_MODEL, D_MODEL, 0);

  // ---- FFN
  layernorm_k<<<M, blk256, 0, stream>>>(x2, ln3_w, ln3_b, n3, D_MODEL);
  gemmL(n3, ffn_w1, ffn_b1, nullptr, ff, M, D_FF, D_MODEL, 1);
  gemmL(ff, ffn_w2, ffn_b2, x2, (float*)d_out, M, D_MODEL, D_FF, 0);
}
